// ScalarHull_11089605558691
// MI455X (gfx1250) — matrix operand order repaired and checked by reading, not compiled, not run on hardware
//
#include <hip/hip_runtime.h>
#include <hip/hip_bf16.h>

// ---- problem constants (match reference) ----
#define D_  1024
#define K_  8
#define P_  8
#define N_  4096
#define DK  8192   // D*K

// ---- GEMM tiling ----
#define BM 128
#define BN 64
#define BK 32

typedef __bf16 v16bf __attribute__((ext_vector_type(16)));
typedef float  v8f   __attribute__((ext_vector_type(8)));
typedef unsigned int v4u __attribute__((ext_vector_type(4)));
typedef int v8i __attribute__((ext_vector_type(8)));
typedef int v4i __attribute__((ext_vector_type(4)));

__device__ __forceinline__ float softplus_f(float x) {
  return fmaxf(x, 0.0f) + log1pf(expf(-fabsf(x)));
}
__device__ __forceinline__ float sigmoid_f(float x) {
  return 1.0f / (1.0f + expf(-x));
}

// ---- CDNA5 async global->LDS copy (ASYNCcnt path), no VGPR staging ----
__device__ __forceinline__ unsigned lds_off(const void* p) {
  // generic LDS pointer: low 32 bits are the LDS byte offset
  return (unsigned)(uintptr_t)p;
}
__device__ __forceinline__ void async_copy_b128(unsigned lds, const void* g) {
  asm volatile("global_load_async_to_lds_b128 %0, %1, off"
               :: "v"(lds), "v"(g)
               : "memory");
}
__device__ __forceinline__ void wait_async0() {
  asm volatile("s_wait_asynccnt 0" ::: "memory");
}

// ---- CDNA5 Tensor Data Mover: one-descriptor 2D tile load (TENSORcnt) ----
// D# per CDNA5 ISA: group0 = {flags, lds_addr, global_addr, type},
// group1 = {data_size, tensor dims/strides, tile dims}. data_size = 8B units,
// tile_dim0 = BK*2/8 = 8 units (one 64B LDS row per tile row).
// This toolchain exposes the 6-arg builtin: (v4u, v8i, v4i, v4i, v8i, i32).
__device__ __forceinline__ void tdm_load_2d(unsigned lds_byte, const void* gaddr,
                                            unsigned tile_rows,
                                            unsigned row_stride_bytes,
                                            unsigned tensor_rows) {
  unsigned long long ga = (unsigned long long)(uintptr_t)gaddr;
  v4u g0;
  g0.x = 1u;                                        // count=1, user descriptor
  g0.y = lds_byte;                                  // lds_addr
  g0.z = (unsigned)(ga & 0xFFFFFFFFu);              // global_addr[31:0]
  g0.w = (unsigned)((ga >> 32) & 0x01FFFFFFu)       // global_addr[56:32]
         | (2u << 30);                              // type = 2 ("image")
  unsigned su = row_stride_bytes >> 3;              // row pitch in 8B units
  v8i g1;
  g1[0] = (int)(3u << 16);                          // data_size = 3 (8 bytes)
  g1[1] = (int)((su & 0xFFFFu) << 16);              // tensor_dim0[15:0] @bits63:48
  g1[2] = (int)(((su >> 16) & 0xFFFFu)              // tensor_dim0[31:16]
         | ((tensor_rows & 0xFFFFu) << 16));        // tensor_dim1[15:0]
  g1[3] = (int)(((tensor_rows >> 16) & 0xFFFFu)     // tensor_dim1[31:16]
         | (8u << 16));                             // tile_dim0 = 8 units (64B)
  g1[4] = (int)(tile_rows & 0xFFFFu);               // tile_dim1 = rows; tile_dim2=0
  g1[5] = (int)su;                                  // tensor_dim0_stride[31:0]
  g1[6] = 0;                                        // stride[47:32], dim1_stride lo
  g1[7] = 0;                                        // dim1_stride hi
  v4i z4 = {0, 0, 0, 0};
  v8i z8 = {0, 0, 0, 0, 0, 0, 0, 0};
  __builtin_amdgcn_tensor_load_to_lds(g0, g1, z4, z4, z8, 0);
}
__device__ __forceinline__ void wait_tensor0() {
  __builtin_amdgcn_s_wait_tensorcnt(0);
}

// -------------------------------------------------------------------
// Weight prep kernels (one pass each; all data stays hot in L2)
// -------------------------------------------------------------------

// wAT[e][r] = bf16(weight_flat[r*D + e]),  r in [0,DK), e in [0,D)
__global__ __launch_bounds__(256) void k_prep_wAT(const float* __restrict__ w,
                                                  __bf16* __restrict__ wAT) {
  int stride = gridDim.x * blockDim.x;
  for (int i = blockIdx.x * blockDim.x + threadIdx.x; i < D_ * DK; i += stride) {
    int e = i >> 13;          // / DK
    int r = i & (DK - 1);
    wAT[i] = (__bf16)w[r * D_ + e];
  }
}

// w2bt[p][e][d] = bf16(softplus(weight2[p][e][d])^2)   (already Bt layout)
__global__ __launch_bounds__(256) void k_prep_w2(const float* __restrict__ w2,
                                                 __bf16* __restrict__ out) {
  int stride = gridDim.x * blockDim.x;
  for (int i = blockIdx.x * blockDim.x + threadIdx.x; i < P_ * D_ * D_; i += stride) {
    float s = softplus_f(w2[i]);
    out[i] = (__bf16)(s * s);
  }
}

// zwT[p][e][d] = bf16(z_weight[p][d][e] + z_weight[p][d+D][e])  (fold concat + transpose)
__global__ __launch_bounds__(256) void k_prep_zw(const float* __restrict__ zin,
                                                 __bf16* __restrict__ out) {
  int stride = gridDim.x * blockDim.x;
  for (int i = blockIdx.x * blockDim.x + threadIdx.x; i < P_ * D_ * D_; i += stride) {
    int p   = i >> 20;               // / (D*D)
    int rem = i & ((1 << 20) - 1);
    int e   = rem >> 10;
    int d   = rem & (D_ - 1);
    const float* zp = zin + p * (2 * D_ * D_);
    out[i] = (__bf16)(zp[d * D_ + e] + zp[(d + D_) * D_ + e]);
  }
}

// -------------------------------------------------------------------
// Row prep: ConvexGate, noise, xg (bf16), tau
// -------------------------------------------------------------------
__global__ __launch_bounds__(256) void k_rowprep(const float* __restrict__ x,
                                                 const float* __restrict__ noise,
                                                 const float* __restrict__ gate_w,
                                                 const float* __restrict__ gate_b,
                                                 __bf16* __restrict__ xg_bf,
                                                 float* __restrict__ tau) {
  __shared__ float red[256];
  int n = blockIdx.x;
  int t = threadIdx.x;
  const float* xr = x + n * D_;
  const float* nr = noise + n * D_;

  float xv[4], nv[4];
  float dot = 0.0f;
#pragma unroll
  for (int i = 0; i < 4; ++i) {
    int d = t + i * 256;
    xv[i] = xr[d];
    nv[i] = nr[d];
    dot += xv[i] * gate_w[d];
  }
  red[t] = dot;
  __syncthreads();
  for (int s = 128; s > 0; s >>= 1) {
    if (t < s) red[t] += red[t + s];
    __syncthreads();
  }
  float g = 1.0f - expf(-softplus_f(red[0] + gate_b[0]));
  __syncthreads();

  float ssq = 0.0f;
#pragma unroll
  for (int i = 0; i < 4; ++i) {
    int d = t + i * 256;
    float v = (xv[i] + nv[i] * 1e-5f) * g;
    ssq += v * v;
    xg_bf[n * D_ + d] = (__bf16)v;
  }
  red[t] = ssq;
  __syncthreads();
  for (int s = 128; s > 0; s >>= 1) {
    if (t < s) red[t] += red[t + s];
    __syncthreads();
  }
  if (t == 0) {
    float r = sqrtf(red[0] * (1.0f / D_) + 1e-6f);
    tau[n] = expf(0.30343f * r + 0.22159f);
  }
}

// -------------------------------------------------------------------
// GEMM1: x_proj = phi(N x DK) @ W(DK x D); phi generated on the fly.
// Double-buffered LDS; weight tiles via async global->LDS (ASYNCcnt).
// Epilogue: z0[p] = bf16(softplus((x_proj+bias) * sigmoid(gate_raw[p])))
// -------------------------------------------------------------------
__global__ __launch_bounds__(256) void k_gemm1(const __bf16* __restrict__ xg,
                                               const __bf16* __restrict__ wAT,
                                               const float* __restrict__ bias,
                                               const float* __restrict__ gate_raw,
                                               __bf16* __restrict__ z0) {
  __shared__ __bf16 As[2][BM * BK];
  __shared__ __bf16 Bs[2][BN * BK];

  int t      = threadIdx.x;
  int lane   = t & 31;
  int wave   = t >> 5;
  int lane16 = lane & 15;
  int lhalf  = lane >> 4;
  int n0     = blockIdx.x * BN;
  int m0     = blockIdx.y * BM;

  v8f acc[4] = {};

  int ar = t >> 2;        // 0..63
  int c0 = (t & 3) * 8;   // 0,8,16,24

  auto fillA = [&](int buf, int kk) {
    for (int rr = ar; rr < BM; rr += 64) {
      int d = (kk + c0) >> 3;  // kk+c0 is a multiple of 8 -> one shared xg value
      float xv = (float)xg[(m0 + rr) * D_ + d];
      union { __bf16 h[8]; uint4 u; } pk;
#pragma unroll
      for (int j = 0; j < 8; ++j) {
        float sh = -1.0f + (2.0f / 7.0f) * (float)j;
        pk.h[j] = (__bf16)softplus_f(xv + sh);
      }
      *reinterpret_cast<uint4*>(&As[buf][rr * BK + c0]) = pk.u;
    }
  };
  auto fillB = [&](int buf, int kk) {
    async_copy_b128(lds_off(&Bs[buf][ar * BK + c0]),
                    &wAT[(n0 + ar) * DK + kk + c0]);
  };

  // prologue
  fillB(0, 0);
  fillA(0, 0);
  wait_async0();
  __syncthreads();

  int cur = 0;
  for (int kk = 0; kk < DK; kk += BK) {
    int nxt = cur ^ 1;
    if (kk + BK < DK) {
      fillB(nxt, kk + BK);   // async: overlaps with WMMA below
      fillA(nxt, kk + BK);   // VALU phi generation overlaps too
    }

    // ---- fragments + WMMA on current buffer ----
    union { uint4 u[2]; v16bf v; } af;
    const __bf16* arp = &As[cur][(wave * 16 + lane16) * BK + lhalf * 8];
    af.u[0] = *reinterpret_cast<const uint4*>(arp);
    af.u[1] = *reinterpret_cast<const uint4*>(arp + 16);
#pragma unroll
    for (int j = 0; j < 4; ++j) {
      union { uint4 u[2]; v16bf v; } bfg;
      const __bf16* brp = &Bs[cur][(j * 16 + lane16) * BK + lhalf * 8];
      bfg.u[0] = *reinterpret_cast<const uint4*>(brp);
      bfg.u[1] = *reinterpret_cast<const uint4*>(brp + 16);
      acc[j] = __builtin_amdgcn_wmma_f32_16x16x32_bf16(
          false, af.v, false, bfg.v, (short)0, acc[j], false, false);
    }

    wait_async0();
    __syncthreads();
    cur = nxt;
  }

  // ---- epilogue: 8 gated softplus copies ----
  float g1[P_];
#pragma unroll
  for (int p = 0; p < P_; ++p) g1[p] = sigmoid_f(gate_raw[p]);

#pragma unroll
  for (int j = 0; j < 4; ++j) {
    int col = n0 + j * 16 + lane16;
    float bcol = bias[col];
#pragma unroll
    for (int i = 0; i < 8; ++i) {
      int row = m0 + wave * 16 + lhalf * 8 + i;
      float v = acc[j][i] + bcol;
#pragma unroll
      for (int p = 0; p < P_; ++p) {
        z0[(p * N_ + row) * D_ + col] = (__bf16)softplus_f(v * g1[p]);
      }
    }
  }
}

// -------------------------------------------------------------------
// Fused GEMM2+GEMM3 + epilogue + row-mean reduction:
//   z1  = z0[p] @ w2bt[p]   (K phase 1: kk in [0,1024))
//   xr  = xg    @ zwT[p]    (K phase 2: kk in [1024,2048))
//   score[n][p] = mean_e softplus(softplus((z1+b2)*g2) + xr) + obias
// Tile traffic via Tensor Data Mover (one descriptor per tile, wave 0),
// double-buffered; completion on TENSORcnt.
// -------------------------------------------------------------------
__global__ __launch_bounds__(256) void k_gemm2_fused(const __bf16* __restrict__ z0,
                                                     const __bf16* __restrict__ xg,
                                                     const __bf16* __restrict__ w2bt,
                                                     const __bf16* __restrict__ zwT,
                                                     const float* __restrict__ bias2,
                                                     const float* __restrict__ obias,
                                                     const float* __restrict__ gate_raw2,
                                                     float* __restrict__ scores) {
  __shared__ __bf16 As[2][BM * BK];
  __shared__ __bf16 Bs[2][BN * BK];

  int t      = threadIdx.x;
  int lane   = t & 31;
  int wave   = t >> 5;
  int lane16 = lane & 15;
  int lhalf  = lane >> 4;
  int p      = blockIdx.x;
  int m0     = blockIdx.y * BM;

  float g2 = sigmoid_f(gate_raw2[p]);
  float part[8];
#pragma unroll
  for (int i = 0; i < 8; ++i) part[i] = 0.0f;

  const __bf16* z0p = z0 + p * (N_ * D_);
  const __bf16* w2p = w2bt + p * (D_ * D_);
  const __bf16* zwp = zwT + p * (D_ * D_);

  // wave 0 / lane 0 issues two TDM descriptors (A tile, B tile) per stage
  auto fillAB = [&](int buf, int kk, int n0) {
    if (t == 0) {
      bool ph1 = (kk < D_);
      int k = ph1 ? kk : kk - D_;
      const __bf16* asrc = ph1 ? &z0p[m0 * D_ + k] : &xg[m0 * D_ + k];
      tdm_load_2d(lds_off(&As[buf][0]), asrc, BM, D_ * 2, N_);
      const __bf16* bsrc = ph1 ? &w2p[n0 * D_ + k] : &zwp[n0 * D_ + k];
      tdm_load_2d(lds_off(&Bs[buf][0]), bsrc, BN, D_ * 2, D_);
    }
  };

  for (int et = 0; et < D_ / BN; ++et) {
    int n0 = et * BN;
    v8f acc1[4] = {};
    v8f acc2[4] = {};

    // prologue for this e-tile
    fillAB(0, 0, n0);
    if (wave == 0) wait_tensor0();
    __syncthreads();

    int cur = 0;
    for (int kk = 0; kk < 2 * D_; kk += BK) {
      int nxt = cur ^ 1;
      if (kk + BK < 2 * D_) fillAB(nxt, kk + BK, n0);  // TDM overlaps with WMMA

      bool ph1 = (kk < D_);
      union { uint4 u[2]; v16bf v; } af;
      const __bf16* arp = &As[cur][(wave * 16 + lane16) * BK + lhalf * 8];
      af.u[0] = *reinterpret_cast<const uint4*>(arp);
      af.u[1] = *reinterpret_cast<const uint4*>(arp + 16);
#pragma unroll
      for (int j = 0; j < 4; ++j) {
        union { uint4 u[2]; v16bf v; } bfg;
        const __bf16* brp = &Bs[cur][(j * 16 + lane16) * BK + lhalf * 8];
        bfg.u[0] = *reinterpret_cast<const uint4*>(brp);
        bfg.u[1] = *reinterpret_cast<const uint4*>(brp + 16);
        if (ph1)
          acc1[j] = __builtin_amdgcn_wmma_f32_16x16x32_bf16(
              false, af.v, false, bfg.v, (short)0, acc1[j], false, false);
        else
          acc2[j] = __builtin_amdgcn_wmma_f32_16x16x32_bf16(
              false, af.v, false, bfg.v, (short)0, acc2[j], false, false);
      }

      if (wave == 0) wait_tensor0();   // issuer drains TENSORcnt, then publish
      __syncthreads();
      cur = nxt;
    }

    // ---- fused epilogue for this e-tile; accumulate row partials ----
#pragma unroll
    for (int j = 0; j < 4; ++j) {
      int e = n0 + j * 16 + lane16;
      float b2 = bias2[p * D_ + e];
      float ob = obias[p * D_ + e];
#pragma unroll
      for (int i = 0; i < 8; ++i) {
        float z1 = acc1[j][i] + b2;
        float v1 = softplus_f(z1 * g2);
        float zf = softplus_f(v1 + acc2[j][i]) + ob;
        part[i] += zf;
      }
    }
  }

  // reduce partials across the 16-lane column group
#pragma unroll
  for (int i = 0; i < 8; ++i) {
    float v = part[i];
    v += __shfl_xor(v, 1, 32);
    v += __shfl_xor(v, 2, 32);
    v += __shfl_xor(v, 4, 32);
    v += __shfl_xor(v, 8, 32);
    part[i] = v;
  }
  if (lane16 == 0) {
#pragma unroll
    for (int i = 0; i < 8; ++i) {
      int row = m0 + wave * 16 + lhalf * 8 + i;
      scores[row * P_ + p] = part[i] * (1.0f / D_);
    }
  }
}

// -------------------------------------------------------------------
// Final: out[n] = logsumexp_p(scores[n][p]*tau[n]) / tau[n]
// -------------------------------------------------------------------
__global__ __launch_bounds__(256) void k_lse(const float* __restrict__ scores,
                                             const float* __restrict__ tau,
                                             float* __restrict__ out) {
  int n = blockIdx.x * 256 + threadIdx.x;
  if (n >= N_) return;
  float tv = tau[n];
  float s[P_];
  float m = -1e30f;
#pragma unroll
  for (int p = 0; p < P_; ++p) {
    s[p] = scores[n * P_ + p] * tv;
    m = fmaxf(m, s[p]);
  }
  float sum = 0.0f;
#pragma unroll
  for (int p = 0; p < P_; ++p) sum += expf(s[p] - m);
  out[n] = (m + logf(sum)) / tv;
}

// -------------------------------------------------------------------
extern "C" void kernel_launch(void* const* d_in, const int* in_sizes, int n_in,
                              void* d_out, int out_size, void* d_ws, size_t ws_size,
                              hipStream_t stream) {
  (void)in_sizes; (void)n_in; (void)out_size; (void)ws_size;
  const float* x           = (const float*)d_in[0];
  const float* noise       = (const float*)d_in[1];
  const float* gate_w      = (const float*)d_in[2];
  const float* gate_b      = (const float*)d_in[3];
  const float* weight      = (const float*)d_in[4];
  const float* bias        = (const float*)d_in[5];
  const float* weight2     = (const float*)d_in[6];
  const float* bias2       = (const float*)d_in[7];
  const float* gate_raw2   = (const float*)d_in[8];
  const float* z_weight    = (const float*)d_in[9];
  const float* gate_raw    = (const float*)d_in[10];
  const float* output_bias = (const float*)d_in[11];
  float* out = (float*)d_out;

  char* ws = (char*)d_ws;
  size_t off = 0;
  auto alloc = [&](size_t bytes) {
    char* pp = ws + off;
    off += (bytes + 255) & ~(size_t)255;
    return pp;
  };
  __bf16* xg_bf  = (__bf16*)alloc((size_t)N_ * D_ * 2);
  float*  tau    = (float*)alloc((size_t)N_ * 4);
  __bf16* wAT    = (__bf16*)alloc((size_t)D_ * DK * 2);
  __bf16* w2bt   = (__bf16*)alloc((size_t)P_ * D_ * D_ * 2);
  __bf16* zwT    = (__bf16*)alloc((size_t)P_ * D_ * D_ * 2);
  __bf16* z0     = (__bf16*)alloc((size_t)P_ * N_ * D_ * 2);
  float*  scores = (float*)alloc((size_t)N_ * P_ * 4);

  k_prep_wAT<<<2048, 256, 0, stream>>>(weight, wAT);
  k_prep_w2<<<2048, 256, 0, stream>>>(weight2, w2bt);
  k_prep_zw<<<2048, 256, 0, stream>>>(z_weight, zwT);
  k_rowprep<<<N_, 256, 0, stream>>>(x, noise, gate_w, gate_b, xg_bf, tau);
  k_gemm1<<<dim3(D_ / BN, N_ / BM), 256, 0, stream>>>(xg_bf, wAT, bias, gate_raw, z0);
  k_gemm2_fused<<<dim3(P_, N_ / BM), 256, 0, stream>>>(z0, xg_bf, w2bt, zwT, bias2,
                                                       output_bias, gate_raw2, scores);
  k_lse<<<(N_ + 255) / 256, 256, 0, stream>>>(scores, tau, out);
}
